// ProbSparseSelfAttention_15942918603180
// MI455X (gfx1250) — matrix operand order repaired and checked by reading, not compiled, not run on hardware
//
#include <hip/hip_runtime.h>
#include <hip/hip_bf16.h>

// Problem constants (match reference: B=2, L=2048, Dm=1024, H=16, D=64, U=38)
#define B_   2
#define H_   16
#define L_   2048
#define D_   64
#define DM_  1024
#define M_   (B_ * L_)   // 4096 rows for the projection GEMMs
#define U_TOP 38         // int(5 * log(2048))

typedef __attribute__((ext_vector_type(16))) __bf16        v16bf;
typedef __attribute__((ext_vector_type(8)))  float         v8f;
typedef __attribute__((ext_vector_type(4)))  unsigned int  u32x4;
typedef __attribute__((ext_vector_type(8)))  unsigned int  u32x8;
typedef __attribute__((ext_vector_type(4)))  float         f32x4;

union Frag {                 // one WMMA 16-bit operand fragment per lane
  v16bf          v;
  u32x4          q[2];       // two 16-byte vector loads
  unsigned short u[16];
};

__device__ __forceinline__ unsigned short f2bf(float f) {
  union { __bf16 h; unsigned short u; } x;
  x.h = (__bf16)f;           // native f32->bf16 cvt
  return x.u;
}

// order-preserving mapping f32 -> u32 (ascending)
__device__ __forceinline__ unsigned fkey(float f) {
  unsigned u = __float_as_uint(f);
  return (u & 0x80000000u) ? ~u : (u | 0x80000000u);
}
__device__ __forceinline__ float fval(unsigned k) {
  return (k & 0x80000000u) ? __uint_as_float(k ^ 0x80000000u)
                           : __uint_as_float(~k);
}

__device__ __forceinline__ v8f wmma_bf16(const Frag& a, const Frag& b, v8f c) {
  return __builtin_amdgcn_wmma_f32_16x16x32_bf16(
      /*neg_a=*/false, a.v, /*neg_b=*/false, b.v,
      /*c_mod=*/(short)0, c, /*reuse_a=*/false, /*reuse_b=*/false);
}

// ---------------------------------------------------------------------------
// One-shot f32 -> bf16 conversion (bandwidth bound; 8 elements/thread)
// ---------------------------------------------------------------------------
__global__ __launch_bounds__(256) void cvt_bf16_kernel(
    const float* __restrict__ src, unsigned short* __restrict__ dst, int n) {
  const int base = (blockIdx.x * 256 + threadIdx.x) * 8;
  if (base + 8 > n) return;
  f32x4 a = *(const f32x4*)(src + base);
  f32x4 b = *(const f32x4*)(src + base + 4);
  u32x4 o;
  o.x = (unsigned)f2bf(a.x) | ((unsigned)f2bf(a.y) << 16);
  o.y = (unsigned)f2bf(a.z) | ((unsigned)f2bf(a.w) << 16);
  o.z = (unsigned)f2bf(b.x) | ((unsigned)f2bf(b.y) << 16);
  o.w = (unsigned)f2bf(b.z) | ((unsigned)f2bf(b.w) << 16);
  *(u32x4*)(dst + base) = o;
}

// ---------------------------------------------------------------------------
// NT GEMM on bf16 operands:  out = A[M,1024] * W[N=1024,K=1024]^T + bias
// One 16x64 macro-tile per wave (A fragment reused across 4 WMMAs per k-step).
// mode 0: out_bf scatter to [B,H,L,D] bf16       (QKV projections)
// mode 1: out_f32 row-major [M, DM] f32          (final output projection)
// ---------------------------------------------------------------------------
__global__ __launch_bounds__(256) void proj_gemm_kernel(
    const unsigned short* __restrict__ A, const unsigned short* __restrict__ W,
    const float* __restrict__ bias, unsigned short* __restrict__ out_bf,
    float* __restrict__ out_f32, int mode) {
  const int wave = threadIdx.x >> 5;
  const int lane = threadIdx.x & 31;
  const int sel  = lane >> 4;
  const int lr   = lane & 15;

  const int tile = blockIdx.x * 8 + wave;     // 4096 macro-tiles total
  const int tm   = tile >> 4;                 // 0..255 (16-row strips)
  const int tn4  = tile & 15;                 // 0..15  (64-col strips)

  const unsigned short* Arow = A + (size_t)(tm * 16 + lr) * DM_;
  const unsigned short* Wrow[4];
  #pragma unroll
  for (int nt = 0; nt < 4; ++nt)
    Wrow[nt] = W + (size_t)(tn4 * 64 + nt * 16 + lr) * DM_;

  v8f zero = {};
  v8f acc[4] = { zero, zero, zero, zero };

  for (int kk = 0; kk < DM_ / 32; ++kk) {
    const int abase = kk * 32 + sel * 16;     // A: K = sel*16 + j
    const int bbase = kk * 32 + sel * 16;     // B: K = sel*16 + j
    __builtin_prefetch(Arow + abase + 64, 0, 3);       // global_prefetch_b8
    Frag a;
    a.q[0] = *(const u32x4*)(Arow + abase);
    a.q[1] = *(const u32x4*)(Arow + abase + 8);
    #pragma unroll
    for (int nt = 0; nt < 4; ++nt) {
      __builtin_prefetch(Wrow[nt] + bbase + 64, 0, 3);
      Frag b;
      b.q[0] = *(const u32x4*)(Wrow[nt] + bbase);
      b.q[1] = *(const u32x4*)(Wrow[nt] + bbase + 8);
      acc[nt] = wmma_bf16(a, b, acc[nt]);
    }
  }

  #pragma unroll
  for (int nt = 0; nt < 4; ++nt) {
    #pragma unroll
    for (int r = 0; r < 8; ++r) {             // C layout: M = r + 8*sel, N = lr
      const int m = tm * 16 + r + 8 * sel;
      const int n = tn4 * 64 + nt * 16 + lr;
      const float v = acc[nt][r] + bias[n];
      if (mode == 0) {
        const int bb = m >> 11, l = m & (L_ - 1);
        const int h  = n >> 6,  d = n & (D_ - 1);
        out_bf[(((size_t)bb * H_ + h) * L_ + l) * D_ + d] = f2bf(v);
      } else {
        out_f32[(size_t)m * DM_ + n] = v;
      }
    }
  }
}

// ---------------------------------------------------------------------------
// ProbSparse attention core on bf16 Q/K/V. One block = (b, h, 16-query tile).
// Phase 2: 8 waves x 256 keys of QK^T.  Phase 6: cooperative P*V where the
// V tiles are staged global->LDS by the Tensor Data Mover (TDM).
// ---------------------------------------------------------------------------
__global__ __launch_bounds__(256) void probsparse_attn_kernel(
    const unsigned short* __restrict__ Q, const unsigned short* __restrict__ K,
    const unsigned short* __restrict__ V, unsigned short* __restrict__ ctx) {
  __shared__ float    sS[16][L_];       // 128 KB score / prob buffer
  __shared__ float    sCtx[16][D_];     // 4 KB context accumulator
  __shared__ __align__(128) unsigned short sVst[2][32 * D_];  // 2x4KB TDM stage
  __shared__ unsigned sLo[16], sHi[16], sCnt[16], sMaxK[16];
  __shared__ float    sSum[16];

  const int tid  = threadIdx.x;
  const int wave = tid >> 5, lane = tid & 31;
  const int sel  = lane >> 4, lr = lane & 15;
  const int qt = blockIdx.x, h = blockIdx.y, b = blockIdx.z;
  const size_t head  = ((size_t)b * H_ + h) * (size_t)L_ * D_;
  const int    qbase = qt * 16;

  // Phase 0: init
  for (int i = tid; i < 16 * D_; i += 256) ((float*)sCtx)[i] = 0.f;
  if (tid < 16) { sLo[tid] = 0u; sHi[tid] = 0xFFFFFFFFu; sMaxK[tid] = 0u; sSum[tid] = 0.f; }

  // Phase 1: Q A-fragments (row M = lr), two k-steps over head dim 64
  Frag aQ[2];
  {
    const unsigned short* qrow = Q + head + (size_t)(qbase + lr) * D_;
    #pragma unroll
    for (int ks = 0; ks < 2; ++ks) {
      const int base = ks * 32 + sel * 16;
      aQ[ks].q[0] = *(const u32x4*)(qrow + base);
      aQ[ks].q[1] = *(const u32x4*)(qrow + base + 8);
    }
  }

  // Phase 2: this wave's 16 key blocks of (Q K^T)/8
  for (int t = 0; t < 16; ++t) {
    const int keybase = wave * 256 + t * 16;
    const unsigned short* krow = K + head + (size_t)(keybase + lr) * D_;
    v8f acc = {};
    #pragma unroll
    for (int ks = 0; ks < 2; ++ks) {
      Frag bk;
      const int base = ks * 32 + sel * 16;
      bk.q[0] = *(const u32x4*)(krow + base);
      bk.q[1] = *(const u32x4*)(krow + base + 8);
      acc = wmma_bf16(aQ[ks], bk, acc);
    }
    #pragma unroll
    for (int r = 0; r < 8; ++r)
      sS[r + 8 * sel][keybase + lr] = acc[r] * 0.125f;   // 1/sqrt(64)
  }
  __syncthreads();

  // Phase 3: per-row U-th largest via 32-step bisection on ordered u32 keys
  const int row = tid >> 4, subl = tid & 15;
  for (int it = 0; it < 32; ++it) {
    if (tid < 16) sCnt[tid] = 0u;
    __syncthreads();
    const unsigned mid = sLo[row] + ((sHi[row] - sLo[row]) >> 1);
    unsigned c = 0;
    for (int j = subl; j < L_; j += 16)
      c += (fkey(sS[row][j]) >= mid) ? 1u : 0u;
    atomicAdd(&sCnt[row], c);
    __syncthreads();
    if (tid < 16) {
      const unsigned m2 = sLo[tid] + ((sHi[tid] - sLo[tid]) >> 1);
      if (sCnt[tid] >= U_TOP) sLo[tid] = m2; else sHi[tid] = m2;
    }
    __syncthreads();
  }

  // Phase 4: row max (max element is always kept)
  {
    unsigned mk = 0;
    for (int j = subl; j < L_; j += 16) {
      const unsigned k = fkey(sS[row][j]);
      mk = (k > mk) ? k : mk;
    }
    atomicMax(&sMaxK[row], mk);
  }
  __syncthreads();

  // Phase 5: mask below threshold, exp, accumulate row sums (in place)
  {
    const unsigned thr = sLo[row];
    const float rmax = fval(sMaxK[row]);
    float s = 0.f;
    for (int j = subl; j < L_; j += 16) {
      const float v = sS[row][j];
      const float p = (fkey(v) >= thr) ? __expf(v - rmax) : 0.f;
      sS[row][j] = p;
      s += p;
    }
    atomicAdd(&sSum[row], s);
  }
  __syncthreads();

  // Phase 6: ctx = P(16 x 2048) * V(2048 x 64), cooperative.
  // wave -> (nt = wave&3: 16-col strip of V, khalf = wave>>2: 1024-key half).
  // Waves 0 and 4 DMA 32x64 bf16 V tiles into sVst[khalf] with the TDM.
  const int nt    = wave & 3;
  const int khalf = wave >> 2;
  const float invs = 1.f / sSum[lr];
  v8f accp = {};
  for (int ks = 0; ks < 32; ++ks) {
    const int kb = khalf * 1024 + ks * 32;
    if ((wave & 3) == 0) {
      // ---- Tensor DMA descriptor (D#), 4-group form, 2D tile 64x32 bf16 ----
      const unsigned long long ga = (unsigned long long)(V + head + (size_t)kb * D_);
      const unsigned lds_addr = (unsigned)(size_t)(void*)&sVst[khalf][0];
      u32x4 g0;
      g0[0] = 1u;                                   // count=1, user descriptor
      g0[1] = lds_addr;                             // lds_addr
      g0[2] = (unsigned)ga;                         // global_addr[31:0]
      g0[3] = (unsigned)((ga >> 32) & 0x01FFFFFFull) | 0x80000000u; // type=2
      u32x8 g1 = {};
      g1[0] = 0x10000u;                             // data_size=1 (2 bytes)
      g1[1] = (unsigned)D_ << 16;                   // tensor_dim0 = 64
      g1[2] = ((unsigned)L_ & 0xFFFFu) << 16;       // tensor_dim1 = 2048
      g1[3] = (unsigned)D_ << 16;                   // tile_dim0 = 64
      g1[4] = 32u;                                  // tile_dim1 = 32 keys
      g1[5] = (unsigned)D_;                         // tensor_dim0_stride = 64
      u32x4 gz = {};                                // groups 2/3: zeros (2D)
      asm volatile("tensor_load_to_lds %0, %1, %2, %3"
                   :: "s"(g0), "s"(g1), "s"(gz), "s"(gz)
                   : "memory");
      asm volatile("s_wait_tensorcnt 0x0" ::: "memory");
    }
    __syncthreads();                 // staged tile visible to all 4 waves
    // A fragment: this wave's probability rows, keys [kb, kb+32)
    Frag ap;
    #pragma unroll
    for (int j = 0; j < 8; ++j) {
      ap.u[j]     = f2bf(sS[lr][kb + sel * 16 + j]     * invs);
      ap.u[j + 8] = f2bf(sS[lr][kb + sel * 16 + 8 + j] * invs);
    }
    // B fragment: staged V tile column nt*16+lr, rows = keys
    Frag bv;
    #pragma unroll
    for (int j = 0; j < 16; ++j)
      bv.u[j] = sVst[khalf][(sel * 16 + j) * D_ + nt * 16 + lr];
    accp = wmma_bf16(ap, bv, accp);
    __syncthreads();                 // all readers done before next DMA
  }
  #pragma unroll
  for (int r = 0; r < 8; ++r)
    atomicAdd(&sCtx[r + 8 * sel][nt * 16 + lr], accp[r]);
  __syncthreads();

  // Phase 7: context in [B, L, Dm] layout (Dm index = h*64 + d), bf16
  for (int i = tid; i < 16 * D_; i += 256) {
    const int m = i >> 6, d = i & 63;
    ctx[((size_t)b * L_ + qbase + m) * DM_ + h * D_ + d] = f2bf(sCtx[m][d]);
  }
}

// ---------------------------------------------------------------------------
extern "C" void kernel_launch(void* const* d_in, const int* in_sizes, int n_in,
                              void* d_out, int out_size, void* d_ws, size_t ws_size,
                              hipStream_t stream) {
  (void)in_sizes; (void)n_in; (void)out_size; (void)ws_size;
  const float* x  = (const float*)d_in[0];
  const float* Wq = (const float*)d_in[1];
  const float* bq = (const float*)d_in[2];
  const float* Wk = (const float*)d_in[3];
  const float* bk = (const float*)d_in[4];
  const float* Wv = (const float*)d_in[5];
  const float* bv = (const float*)d_in[6];
  const float* Wo = (const float*)d_in[7];
  const float* bo = (const float*)d_in[8];
  float* out = (float*)d_out;

  // bf16 workspace layout (ushort elements)
  const size_t nx = (size_t)M_ * DM_;        // 4M  (x, ctx, and each of Q/K/V)
  const size_t nw = (size_t)DM_ * DM_;       // 1M  (each weight)
  unsigned short* xb  = (unsigned short*)d_ws;
  unsigned short* wqb = xb  + nx;
  unsigned short* wkb = wqb + nw;
  unsigned short* wvb = wkb + nw;
  unsigned short* wob = wvb + nw;
  unsigned short* Qb  = wob + nw;
  unsigned short* Kb  = Qb  + nx;
  unsigned short* Vb  = Kb  + nx;
  unsigned short* Cxb = Vb  + nx;            // total 24M ushorts = 48 MB

  const dim3 blk(256);

  // Stage 0: one-shot f32 -> bf16 of activations and weights
  cvt_bf16_kernel<<<(int)(nx / (256 * 8)), blk, 0, stream>>>(x,  xb,  (int)nx);
  cvt_bf16_kernel<<<(int)(nw / (256 * 8)), blk, 0, stream>>>(Wq, wqb, (int)nw);
  cvt_bf16_kernel<<<(int)(nw / (256 * 8)), blk, 0, stream>>>(Wk, wkb, (int)nw);
  cvt_bf16_kernel<<<(int)(nw / (256 * 8)), blk, 0, stream>>>(Wv, wvb, (int)nw);
  cvt_bf16_kernel<<<(int)(nw / (256 * 8)), blk, 0, stream>>>(Wo, wob, (int)nw);

  // Stage 1: Q/K/V projections (bf16 in, bf16 out in [B,H,L,D])
  const int gemm_blocks = (M_ / 16) * (DM_ / 64) / 8;   // 512
  proj_gemm_kernel<<<gemm_blocks, blk, 0, stream>>>(xb, wqb, bq, Qb, nullptr, 0);
  proj_gemm_kernel<<<gemm_blocks, blk, 0, stream>>>(xb, wkb, bk, Kb, nullptr, 0);
  proj_gemm_kernel<<<gemm_blocks, blk, 0, stream>>>(xb, wvb, bv, Vb, nullptr, 0);

  // Stage 2: ProbSparse attention -> bf16 context in [B, L, Dm]
  probsparse_attn_kernel<<<dim3(L_ / 16, H_, B_), blk, 0, stream>>>(Qb, Kb, Vb, Cxb);

  // Stage 3: output projection -> f32 result
  proj_gemm_kernel<<<gemm_blocks, blk, 0, stream>>>(Cxb, wob, bo, nullptr, out, 1);
}
